// SL_GAD_Model_43808666419361
// MI455X (gfx1250) — matrix operand order repaired and checked by reading, not compiled, not run on hardware
//
#include <hip/hip_runtime.h>
#include <math.h>

// ---------------------------------------------------------------------------
// SL-GAD forward, MI455X (gfx1250, wave32, WMMA f16 -> f32 accum)
// ---------------------------------------------------------------------------

typedef _Float16 f16;
typedef __attribute__((ext_vector_type(16))) _Float16 v16h;
typedef __attribute__((ext_vector_type(8)))  float    v8f;

#define B_    8192
#define K_    16
#define E_    64
#define FIN_  300
#define FOUT_ 64
#define ALPHA_ 1.0f
#define BETA_  0.6f
#define EPS_   1e-12f

#define ENC_WAVES 4
#define YSTR 68            // 16x64 tile padded to stride 68 (bank-friendly)
#define ENC_WAVE_LDS 2384  // Y(1088) + H(1088) + EW(64) + ES(64) + ED(64) + NS(16)

#define DEC_WAVES 2
#define NPAD 304           // 300 padded to 19 tiles of 16
#define HSTR 312

__device__ __forceinline__ float wave_sum(float v) {
#pragma unroll
  for (int off = 16; off > 0; off >>= 1) v += __shfl_xor(v, off, 32);
  return v;
}

// ---------------------------------------------------------------------------
// prep: zero scalar accumulators; build f16 transposed weights
//   wtenc[n][k] (64 x 320), wtdec[n][k] (304 x 64), zero-padded
// ---------------------------------------------------------------------------
__global__ void slgad_prep(const float* __restrict__ W_enc,
                           const float* __restrict__ W_dec,
                           float* __restrict__ scal,
                           f16* __restrict__ wtenc,
                           f16* __restrict__ wtdec) {
  int tid = blockIdx.x * blockDim.x + threadIdx.x;
  int nth = gridDim.x * blockDim.x;
  if (tid < 8) scal[tid] = 0.f;
  for (int i = tid; i < 64 * 320; i += nth) {
    int n = i / 320, k = i % 320;
    wtenc[i] = (k < FIN_) ? (f16)W_enc[k * FOUT_ + n] : (f16)0.f;
  }
  for (int i = tid; i < NPAD * 64; i += nth) {
    int n = i / 64, k = i % 64;
    wtdec[i] = (n < FIN_) ? (f16)W_dec[k * FIN_ + n] : (f16)0.f;
  }
}

// ---------------------------------------------------------------------------
// enc: one wave per batch graph.
//   Y = feat[b] (16x300) @ W_enc (300x64) via v_wmma_f32_16x16x32_f16
//   anchor, spmm->h, pool, per-row L2 norms, and fused dec-prep:
//   v[b] = sum_{e:dst=0} w_e * nsrc[src_e] * h_l2n[src_e]   (64-dim)
// ---------------------------------------------------------------------------
__global__ __launch_bounds__(ENC_WAVES * 32)
void slgad_enc(const float* __restrict__ feat0, const float* __restrict__ feat1,
               const float* __restrict__ feat2,
               const float* __restrict__ w0, const float* __restrict__ w1,
               const float* __restrict__ w2,
               const int* __restrict__ src0, const int* __restrict__ src1,
               const int* __restrict__ src2,
               const int* __restrict__ dst0, const int* __restrict__ dst1,
               const int* __restrict__ dst2,
               const float* __restrict__ benc, const f16* __restrict__ wtenc,
               float* __restrict__ pool_out,   // 3*B*64
               float* __restrict__ anch_out,   // 2*B*64
               float* __restrict__ v_out,      // 2*B*64
               float* __restrict__ nd_out)     // 2*B
{
  __shared__ float smem[ENC_WAVES * ENC_WAVE_LDS];
  const int lane = threadIdx.x & 31;
  const int wave = threadIdx.x >> 5;
  const int pe = blockIdx.y;
  const int b  = blockIdx.x * ENC_WAVES + wave;

  const float* feat = (pe == 0) ? feat0 : (pe == 1) ? feat1 : feat2;
  const float* wgt  = (pe == 0) ? w0    : (pe == 1) ? w1    : w2;
  const int*   src  = (pe == 0) ? src0  : (pe == 1) ? src1  : src2;
  const int*   dst  = (pe == 0) ? dst0  : (pe == 1) ? dst1  : dst2;

  float* Y  = smem + wave * ENC_WAVE_LDS;  // 16 x YSTR
  float* H  = Y + 16 * YSTR;               // 16 x YSTR
  float* EW = H + 16 * YSTR;               // 64 edge weights
  int*   ES = (int*)(EW + 64);             // 64 src
  int*   ED = ES + 64;                     // 64 dst
  float* NS = (float*)(ED + 64);           // 16 deg_inv_sqrt(src)

  // stage edges (wave-uniform reads later)
  {
    int e0 = lane, e1 = lane + 32;
    EW[e0] = wgt[(size_t)b * E_ + e0]; EW[e1] = wgt[(size_t)b * E_ + e1];
    ES[e0] = src[(size_t)b * E_ + e0]; ES[e1] = src[(size_t)b * E_ + e1];
    ED[e0] = dst[(size_t)b * E_ + e0]; ED[e1] = dst[(size_t)b * E_ + e1];
  }

  // ---- GEMM: 16x300 @ 300x64, K padded to 320, f16 WMMA ----
  v8f c0 = {}, c1 = {}, c2 = {}, c3 = {};
  const int rowA = lane & 15;
  const int kbA  = (lane < 16) ? 0 : 16;   // A-fragment K swizzle
  const int kbB  = (lane < 16) ? 0 : 16;   // B-fragment K swizzle
  const int colB = lane & 15;
  const float* frow = feat + ((size_t)b * K_ + rowA) * FIN_;

  for (int kk = 0; kk < 320; kk += 32) {
    v16h A;
#pragma unroll
    for (int i = 0; i < 8; i++) {
      int k0 = kk + kbA + i;
      int k1 = kk + kbA + 8 + i;
      A[i]     = (k0 < FIN_) ? (f16)frow[k0] : (f16)0.f;
      A[8 + i] = (k1 < FIN_) ? (f16)frow[k1] : (f16)0.f;
    }
    const f16* wb = wtenc + kk + kbB;
    v16h Bf0 = *(const v16h*)(wb + (size_t)(colB +  0) * 320);
    v16h Bf1 = *(const v16h*)(wb + (size_t)(colB + 16) * 320);
    v16h Bf2 = *(const v16h*)(wb + (size_t)(colB + 32) * 320);
    v16h Bf3 = *(const v16h*)(wb + (size_t)(colB + 48) * 320);
    c0 = __builtin_amdgcn_wmma_f32_16x16x32_f16(false, A, false, Bf0, (short)0, c0, false, false);
    c1 = __builtin_amdgcn_wmma_f32_16x16x32_f16(false, A, false, Bf1, (short)0, c1, false, false);
    c2 = __builtin_amdgcn_wmma_f32_16x16x32_f16(false, A, false, Bf2, (short)0, c2, false, false);
    c3 = __builtin_amdgcn_wmma_f32_16x16x32_f16(false, A, false, Bf3, (short)0, c3, false, false);
  }

  // C layout: VGPR r -> row (r + 0/8), col = lane&15 (+16 per tile)
#pragma unroll
  for (int r = 0; r < 8; r++) {
    int row = r + ((lane < 16) ? 0 : 8);
    Y[row * YSTR + colB     ] = c0[r];
    Y[row * YSTR + colB + 16] = c1[r];
    Y[row * YSTR + colB + 32] = c2[r];
    Y[row * YSTR + colB + 48] = c3[r];
  }
  __syncthreads();

  const float be0 = benc[lane], be1 = benc[lane + 32];

  // anchor_out = l2n(relu(feat[:,0,:] @ W + b)) -- only needed for p1/p2
  if (pe < 2) {
    float a0 = fmaxf(Y[lane]      + be0, 0.f);
    float a1 = fmaxf(Y[lane + 32] + be1, 0.f);
    float nrm = sqrtf(wave_sum(a0 * a0 + a1 * a1));
    float inv = 1.f / fmaxf(nrm, EPS_);
    float* ao = anch_out + ((size_t)pe * B_ + b) * FOUT_;
    ao[lane] = a0 * inv; ao[lane + 32] = a1 * inv;
  }

  // spmm: h_pre[d] += w_e * x[s]   (x row 0 is zero)
  for (int i = lane; i < 16 * YSTR; i += 32) H[i] = 0.f;
  __syncthreads();
  for (int e = 0; e < E_; e++) {
    int s = ES[e];
    if (s != 0) {               // x[0,:] == 0
      float we = EW[e];
      int d = ED[e];
      H[d * YSTR + lane]      += we * Y[s * YSTR + lane];
      H[d * YSTR + lane + 32] += we * Y[s * YSTR + lane + 32];
    }
  }

  // h = relu(agg + b); pool = mean_k h; then L2-normalize each row in place
  float pa0 = 0.f, pa1 = 0.f;
  for (int k = 0; k < K_; k++) {
    float h0 = fmaxf(H[k * YSTR + lane]      + be0, 0.f);
    float h1 = fmaxf(H[k * YSTR + lane + 32] + be1, 0.f);
    pa0 += h0; pa1 += h1;
    float nrm = sqrtf(wave_sum(h0 * h0 + h1 * h1));
    float inv = 1.f / fmaxf(nrm, EPS_);
    H[k * YSTR + lane]      = h0 * inv;
    H[k * YSTR + lane + 32] = h1 * inv;
  }
  pa0 *= (1.f / 16.f); pa1 *= (1.f / 16.f);
  {
    float nrm = sqrtf(wave_sum(pa0 * pa0 + pa1 * pa1));
    float inv = 1.f / fmaxf(nrm, EPS_);
    float* po = pool_out + ((size_t)pe * B_ + b) * FOUT_;
    po[lane] = pa0 * inv; po[lane + 32] = pa1 * inv;
  }

  // ---- fused decoder prep (only hd[:,0,:] is consumed downstream) ----
  if (pe < 2) {
    if (lane < 16) {
      int c = 0;
      for (int e = 0; e < E_; e++) c += (ES[e] == lane);
      NS[lane] = 1.f / sqrtf((float)((c < 1) ? 1 : c));
    }
    if (lane == 0) {
      int c = 0;
      for (int e = 0; e < E_; e++) c += (ED[e] == 0);
      nd_out[(size_t)pe * B_ + b] = 1.f / sqrtf((float)((c < 1) ? 1 : c));
    }
    __syncthreads();
    float v0 = 0.f, v1 = 0.f;
    for (int e = 0; e < E_; e++) {
      if (ED[e] == 0) {
        int s = ES[e];
        float f = EW[e] * NS[s];
        v0 += f * H[s * YSTR + lane];
        v1 += f * H[s * YSTR + lane + 32];
      }
    }
    float* vo = v_out + ((size_t)pe * B_ + b) * FOUT_;
    vo[lane] = v0; vo[lane + 32] = v1;
  }
}

// ---------------------------------------------------------------------------
// dec: hd0 = relu((v @ W_dec) * ndst0 + b_dec); da = l2n(hd0);
//      accumulate S = sum((da - ori)^2)  (feeds both mean-MSE and Frobenius)
//   GEMM (B x 64) @ (64 x 304): one wave per 16-batch M-tile, 19 N-tiles
// ---------------------------------------------------------------------------
__global__ __launch_bounds__(DEC_WAVES * 32)
void slgad_dec(const float* __restrict__ feat0, const float* __restrict__ feat1,
               const float* __restrict__ v_in, const float* __restrict__ nd_in,
               const float* __restrict__ bdec, const f16* __restrict__ wtdec,
               float* __restrict__ scal) {
  __shared__ float smem[DEC_WAVES * 16 * HSTR];
  const int lane = threadIdx.x & 31;
  const int wave = threadIdx.x >> 5;
  const int pe = blockIdx.y;
  const int btile = (blockIdx.x * DEC_WAVES + wave) * 16;

  const float* feat = (pe == 0) ? feat0 : feat1;
  const float* V  = v_in + (size_t)pe * B_ * FOUT_;
  const float* ND = nd_in + (size_t)pe * B_;
  float* HD = smem + wave * 16 * HSTR;

  const int rowA = lane & 15;
  const int kbA  = (lane < 16) ? 0 : 16;
  const int kbB  = (lane < 16) ? 0 : 16;
  const int colB = lane & 15;
  const float* vrow = V + (size_t)(btile + rowA) * FOUT_;

  v16h A0, A1;   // K = 0..31 and 32..63
#pragma unroll
  for (int i = 0; i < 8; i++) {
    A0[i]     = (f16)vrow[kbA + i];
    A0[8 + i] = (f16)vrow[kbA + 8 + i];
    A1[i]     = (f16)vrow[32 + kbA + i];
    A1[8 + i] = (f16)vrow[32 + kbA + 8 + i];
  }
  float ndv[8];
#pragma unroll
  for (int r = 0; r < 8; r++) ndv[r] = ND[btile + r + ((lane < 16) ? 0 : 8)];

  for (int nt = 0; nt < 19; nt++) {
    const f16* wb = wtdec + (size_t)(nt * 16 + colB) * 64;
    v16h B0 = *(const v16h*)(wb + kbB);
    v16h B1 = *(const v16h*)(wb + 32 + kbB);
    v8f c = {};
    c = __builtin_amdgcn_wmma_f32_16x16x32_f16(false, A0, false, B0, (short)0, c, false, false);
    c = __builtin_amdgcn_wmma_f32_16x16x32_f16(false, A1, false, B1, (short)0, c, false, false);
    int col = nt * 16 + colB;
    float bd = (col < FIN_) ? bdec[col] : 0.f;
#pragma unroll
    for (int r = 0; r < 8; r++) {
      int row = r + ((lane < 16) ? 0 : 8);
      float hv = fmaxf(c[r] * ndv[r] + bd, 0.f);
      HD[row * HSTR + col] = (col < FIN_) ? hv : 0.f;
    }
  }
  __syncthreads();

  float sacc = 0.f;
  for (int row = 0; row < 16; row++) {
    float ss = 0.f;
    for (int c2 = lane; c2 < NPAD; c2 += 32) {
      float x = HD[row * HSTR + c2];
      ss += x * x;
    }
    float nrm = sqrtf(wave_sum(ss));
    float inv = 1.f / fmaxf(nrm, EPS_);
    const float* orow = feat + (size_t)(btile + row) * K_ * FIN_;  // feat[:,0,:]
    for (int c2 = lane; c2 < FIN_; c2 += 32) {
      float d = HD[row * HSTR + c2] * inv - orow[c2];
      sacc += d * d;
    }
  }
  sacc = wave_sum(sacc);
  if (lane == 0) atomicAdd(scal + 2 + pe, sacc);
}

// ---------------------------------------------------------------------------
// disc: ps/ns = sigmoid(pool^T W_bil a + b); contr[b]; atomics into m1/m2
// ---------------------------------------------------------------------------
__global__ __launch_bounds__(256)
void slgad_disc(const float* __restrict__ Wbil, const float* __restrict__ bbil,
                const float* __restrict__ pool_in, const float* __restrict__ anch_in,
                float* __restrict__ contr, float* __restrict__ scal) {
  __shared__ float Wl[64 * 64];
  const int lane = threadIdx.x & 31;
  const int wave = threadIdx.x >> 5;
  for (int i = threadIdx.x; i < 64 * 64; i += 256) Wl[i] = Wbil[i];
  __syncthreads();

  const int b = blockIdx.x * 8 + wave;
  const float* a1 = anch_in + (size_t)b * 64;
  const float* a2 = anch_in + ((size_t)B_ + b) * 64;
  const float* p1 = pool_in + (size_t)b * 64;
  const float* p2 = pool_in + ((size_t)B_ + b) * 64;
  const float* pn = pool_in + ((size_t)2 * B_ + b) * 64;

  const int j0 = lane, j1 = lane + 32;
  float u1a = 0.f, u1b = 0.f, u2a = 0.f, u2b = 0.f;
  for (int k = 0; k < 64; k++) {
    float a1k = a1[k], a2k = a2[k];
    u1a += Wl[j0 * 64 + k] * a1k;  u1b += Wl[j1 * 64 + k] * a1k;
    u2a += Wl[j0 * 64 + k] * a2k;  u2b += Wl[j1 * 64 + k] * a2k;
  }
  float ps1 = wave_sum(p1[j0] * u1a + p1[j1] * u1b);
  float ns1 = wave_sum(pn[j0] * u1a + pn[j1] * u1b);
  float ps2 = wave_sum(p2[j0] * u2a + p2[j1] * u2b);
  float ns2 = wave_sum(pn[j0] * u2a + pn[j1] * u2b);
  if (lane == 0) {
    float bb = bbil[0];
    float sp1 = 1.f / (1.f + expf(-(ps1 + bb)));
    float sn1 = 1.f / (1.f + expf(-(ns1 + bb)));
    float sp2 = 1.f / (1.f + expf(-(ps2 + bb)));
    float sn2 = 1.f / (1.f + expf(-(ns2 + bb)));
    contr[b] = 0.5f * (sn1 - sp1 + 1.f) + 0.5f * (sn2 - sp2 + 1.f);
    atomicAdd(scal + 0, logf(sp1) + logf(1.f - sn1));
    atomicAdd(scal + 1, logf(sp2) + logf(1.f - sn2));
  }
}

// ---------------------------------------------------------------------------
// final: L and per-batch scores
// ---------------------------------------------------------------------------
__global__ void slgad_final(const float* __restrict__ scal,
                            const float* __restrict__ contr,
                            float* __restrict__ out) {
  int tid = blockIdx.x * blockDim.x + threadIdx.x;
  float S1 = scal[2], S2 = scal[3];
  float gen = 0.5f * (sqrtf(S1) + sqrtf(S2)) / sqrtf((float)FIN_);
  if (tid < B_) out[1 + tid] = ALPHA_ * contr[tid] + BETA_ * gen;
  if (tid == 0) {
    float m1 = scal[0] / (float)B_, m2 = scal[1] / (float)B_;
    float Lcon = -0.25f * (m1 + m2);
    float Lgen = 0.5f * (S1 + S2) / ((float)B_ * (float)FIN_);
    out[0] = ALPHA_ * Lcon + BETA_ * Lgen;
  }
}

// ---------------------------------------------------------------------------
// launch
// ---------------------------------------------------------------------------
extern "C" void kernel_launch(void* const* d_in, const int* in_sizes, int n_in,
                              void* d_out, int out_size, void* d_ws, size_t ws_size,
                              hipStream_t stream) {
  const float* feat_p1 = (const float*)d_in[0];
  const float* feat_p2 = (const float*)d_in[1];
  const float* feat_n  = (const float*)d_in[2];
  const float* w_p1    = (const float*)d_in[3];
  const float* w_p2    = (const float*)d_in[4];
  const float* w_n     = (const float*)d_in[5];
  const float* W_enc   = (const float*)d_in[6];
  const float* b_enc   = (const float*)d_in[7];
  const float* W_dec   = (const float*)d_in[8];
  const float* b_dec   = (const float*)d_in[9];
  const float* W_bil   = (const float*)d_in[10];
  const float* b_bil   = (const float*)d_in[11];
  const int* src_p1 = (const int*)d_in[12];
  const int* dst_p1 = (const int*)d_in[13];
  const int* src_p2 = (const int*)d_in[14];
  const int* dst_p2 = (const int*)d_in[15];
  const int* src_n  = (const int*)d_in[16];
  const int* dst_n  = (const int*)d_in[17];

  // workspace layout (floats); total ~14.9 MB
  float* ws = (float*)d_ws;
  float* scal  = ws + 0;                        // 8
  float* contr = ws + 8;                        // B
  float* pool  = ws + 8200;                     // 3*B*64
  float* anch  = pool + 3 * (size_t)B_ * 64;    // 2*B*64
  float* vv    = anch + 2 * (size_t)B_ * 64;    // 2*B*64
  float* nd    = vv   + 2 * (size_t)B_ * 64;    // 2*B
  f16* wtenc = (f16*)(nd + 2 * (size_t)B_);     // 64*320 halfs
  f16* wtdec = wtenc + 64 * 320;                // 304*64 halfs

  slgad_prep<<<64, 256, 0, stream>>>(W_enc, W_dec, scal, wtenc, wtdec);

  slgad_enc<<<dim3(B_ / ENC_WAVES, 3), ENC_WAVES * 32, 0, stream>>>(
      feat_p1, feat_p2, feat_n, w_p1, w_p2, w_n,
      src_p1, src_p2, src_n, dst_p1, dst_p2, dst_n,
      b_enc, wtenc, pool, anch, vv, nd);

  slgad_disc<<<B_ / 8, 256, 0, stream>>>(W_bil, b_bil, pool, anch, contr, scal);

  slgad_dec<<<dim3(B_ / (16 * DEC_WAVES), 2), DEC_WAVES * 32, 0, stream>>>(
      feat_p1, feat_p2, vv, nd, b_dec, wtdec, scal);

  slgad_final<<<(B_ + 255) / 256, 256, 0, stream>>>(scal, contr, (float*)d_out);
}
